// NEURONSv2LRA_78211354460526
// MI455X (gfx1250) — matrix operand order repaired and checked by reading, not compiled, not run on hardware
//
#include <hip/hip_runtime.h>

// ---------------------------------------------------------------------------
// Fused: emb gather -> block attention (A^T GEMM) -> 2x (GEMM + tanh) ->
//        column partial sums -> (finalize kernel) pool + 256->2 head.
// All matrix math via v_wmma_f32_16x16x32_bf16 (CDNA5 / gfx1250, wave32).
// tanh via hardware V_TANH_F32; intermediate bf16 stores use truncation so
// they lower to a bare high-half 16-bit store (ds_store_b16_d16_hi pattern).
// ---------------------------------------------------------------------------

typedef __attribute__((ext_vector_type(16))) __bf16 v16bf;
typedef __attribute__((ext_vector_type(8)))  float  v8f;

#define S_LEN    4096
#define DIM      256
#define NWIN     64
#define NBATCH   32
#define GAMMA_HZ 60.0f
#define PI_F     3.14159265358979323846f

// workspace layout (in 32-bit dwords)
#define A_OFF   0            //  64 windows * 2048 dwords (A^T bf16 panels) = 512 KB
#define W1_OFF  131072       //  32768 dwords (W1 bf16 B-panel)            = 128 KB
#define W2_OFF  163840       //  32768 dwords (W2 bf16 B-panel)            = 128 KB
#define P_OFF   196608       //  2048 * 256 floats (per-block partial sums) = 2 MB

#if __has_builtin(__builtin_amdgcn_tanhf)
__device__ __forceinline__ float fast_tanh(float x) { return __builtin_amdgcn_tanhf(x); }
#else
__device__ __forceinline__ float fast_tanh(float x) { return tanhf(x); }
#endif

// RNE conversion: used in one-time prep kernels (weights / attention panels).
__device__ __forceinline__ unsigned short f2bf_rne(float x) {
  unsigned u = __builtin_bit_cast(unsigned, x);
  u += 0x7FFFu + ((u >> 16) & 1u);
  return (unsigned short)(u >> 16);
}
__device__ __forceinline__ unsigned pack2(float a, float b) {
  return (unsigned)f2bf_rne(a) | ((unsigned)f2bf_rne(b) << 16);
}
// Truncating conversion: hot path (intermediate activations). Lowers to a
// single high-half move / d16_hi store — no rounding ALU ops.
__device__ __forceinline__ unsigned short f2bf_tr(float x) {
  return (unsigned short)(__builtin_bit_cast(unsigned, x) >> 16);
}

union Frag16 { uint4 q[2]; v16bf v; };

__device__ __forceinline__ v8f vzero8() {
  v8f z;
#pragma unroll
  for (int i = 0; i < 8; ++i) z[i] = 0.0f;
  return z;
}

__device__ __forceinline__ v16bf load_frag_global(const unsigned* __restrict__ p) {
  Frag16 f;
  f.q[0] = ((const uint4*)p)[0];
  f.q[1] = ((const uint4*)p)[1];
  return f.v;
}

// ---------------------------------------------------------------------------
// Prep 1: per-window attention matrices A^T in WMMA A-fragment panel layout.
//   A[s][t] = exp(cos(ph_s - ph_t)) / sum_t' exp(cos(ph_s - ph_t'))
//   GEMM uses A^T[t][s] (M=t rows, K=s).  64 blocks x 256 threads.
// ---------------------------------------------------------------------------
__global__ __launch_bounds__(256) void prep_a_kernel(unsigned* __restrict__ Ap) {
  __shared__ float ph[64];
  __shared__ float rs[64];
  const int w = blockIdx.x, tid = threadIdx.x;
  if (tid < 64) {
    const int g = w * 64 + tid;
    ph[tid] = GAMMA_HZ * 2.0f * PI_F * ((float)g / (float)(S_LEN - 1));
  }
  __syncthreads();
  if (tid < 64) {
    const float p = ph[tid];
    float s = 0.0f;
    for (int t = 0; t < 64; ++t) s += expf(cosf(p - ph[t]));
    rs[tid] = s;
  }
  __syncthreads();
  // tid = (kk*4 + mi)*32 + lane ; A-frag 16x32 bf16 per (mi,kk)
  const int f = tid >> 5, l = tid & 31;
  const int kk = f >> 2, mi = f & 3;
  const int t = mi * 16 + (l & 15);
  const float pt = ph[t];
  unsigned* dst = Ap + (size_t)w * 2048 + (size_t)tid * 8;
#pragma unroll
  for (int j = 0; j < 8; ++j) {
    const int klocal = ((l & 16) ? 16 : 0) + 2 * j;
    const int s0 = kk * 32 + klocal;
    const float v0 = expf(cosf(ph[s0] - pt)) / rs[s0];
    const float v1 = expf(cosf(ph[s0 + 1] - pt)) / rs[s0 + 1];
    dst[j] = pack2(v0, v1);
  }
}

// ---------------------------------------------------------------------------
// Prep 2: repack W (DxD f32, row n = output, col d = input) into bf16
// WMMA B-fragment panels: B[k=d][n] = W[n][d].  128 blocks x 32 threads.
// ---------------------------------------------------------------------------
__global__ __launch_bounds__(32) void prep_w_kernel(const float* __restrict__ W,
                                                    unsigned* __restrict__ Wp) {
  const int bi = blockIdx.x;            // kk*16 + ni
  const int l  = threadIdx.x;
  const int kk = bi >> 4, ni = bi & 15;
  const int n  = ni * 16 + (l & 15);
  const int kb = kk * 32 + ((l & 16) ? 16 : 0);
  unsigned* dst = Wp + (size_t)(((kk * 16 + ni) * 32 + l)) * 8;
  const float* src = W + (size_t)n * DIM + kb;
#pragma unroll
  for (int j = 0; j < 8; ++j) dst[j] = pack2(src[2 * j], src[2 * j + 1]);
}

// ---------------------------------------------------------------------------
// Main fused kernel: one block per (batch, window).  8 waves x 32 lanes.
// ---------------------------------------------------------------------------
__global__ __launch_bounds__(256) void
fused_attn_mlp(const int* __restrict__ x, const float* __restrict__ emb,
               const float* __restrict__ b1, const float* __restrict__ b2,
               const unsigned* __restrict__ Ap, const unsigned* __restrict__ W1p,
               const unsigned* __restrict__ W2p, float* __restrict__ partial) {
  __shared__ __align__(16) unsigned s_bpan[8192];   // 32 KB: H0 B-panel / colsum alias
  __shared__ __align__(16) unsigned s_hrow[8192];   // 32 KB: 64 x 256 bf16 row-major

  const int tid  = threadIdx.x;
  const int lane = tid & 31;
  const int wv   = tid >> 5;              // wave id 0..7
  const int blk  = blockIdx.x;            // b*64 + w
  const int bb   = blk >> 6;
  const int ww   = blk & 63;

  // ---- 1. gather embeddings straight into bf16 B-panel layout in LDS ----
  {
    const int tokbase = bb * S_LEN + ww * 64;
#pragma unroll 4
    for (int it = 0; it < 32; ++it) {
      const int s0 = it * 2;                              // token pair (s0, s0+1)
      const int r0 = x[tokbase + s0];
      const int r1 = x[tokbase + s0 + 1];
      const float e0 = emb[(size_t)r0 * DIM + tid];
      const float e1 = emb[(size_t)r1 * DIM + tid];
      const int kk = s0 >> 5;
      const int sk = s0 & 31;
      const int fl = (tid & 15) | (sk & 16);              // fragment lane
      const int j  = (sk & 15) >> 1;                      // fragment dword
      const int ni = tid >> 4;                            // n-tile
      s_bpan[((((kk << 4) + ni) << 5) + fl) * 8 + j] = pack2(e0, e1);
    }
  }
  __syncthreads();

  const int nj0  = wv * 2;                // this wave owns n-tiles nj0, nj0+1
  const int cn   = lane & 15;             // column within n-tile
  const int rsel = (lane >> 4) << 3;      // 0 or 8 (row offset inside m-tile)

  v8f acc[8];

  // ---- 2. GEMM1: H1(64x256) = A^T(64x64) @ H0(64x256) ----
#pragma unroll
  for (int t = 0; t < 8; ++t) acc[t] = vzero8();
#pragma unroll
  for (int kk = 0; kk < 2; ++kk) {
    Frag16 fb0, fb1;
    {
      const int base0 = ((((kk << 4) + nj0) << 5) + lane) * 8;
      const int base1 = ((((kk << 4) + nj0 + 1) << 5) + lane) * 8;
      fb0.q[0] = *(const uint4*)&s_bpan[base0];
      fb0.q[1] = *(const uint4*)&s_bpan[base0 + 4];
      fb1.q[0] = *(const uint4*)&s_bpan[base1];
      fb1.q[1] = *(const uint4*)&s_bpan[base1 + 4];
    }
#pragma unroll
    for (int mi = 0; mi < 4; ++mi) {
      const unsigned* ap = Ap + ((size_t)ww << 11) + (size_t)(((((kk << 2) + mi) << 5) + lane)) * 8;
      v16bf fa = load_frag_global(ap);
      acc[mi * 2 + 0] = __builtin_amdgcn_wmma_f32_16x16x32_bf16(
          false, fa, false, fb0.v, (short)0, acc[mi * 2 + 0], false, false);
      acc[mi * 2 + 1] = __builtin_amdgcn_wmma_f32_16x16x32_bf16(
          false, fa, false, fb1.v, (short)0, acc[mi * 2 + 1], false, false);
    }
  }
  __syncthreads();   // all B-panel reads complete

  // store H1 (bf16 row-major, truncating convert -> bare 16-bit hi stores)
  {
    unsigned short* hb = (unsigned short*)s_hrow;
#pragma unroll
    for (int mi = 0; mi < 4; ++mi)
#pragma unroll
      for (int nj = 0; nj < 2; ++nj) {
        const int n  = (nj0 + nj) * 16 + cn;
        const int rb = mi * 16 + rsel;
        v8f a = acc[mi * 2 + nj];
#pragma unroll
        for (int v = 0; v < 8; ++v) hb[(rb + v) * DIM + n] = f2bf_tr(a[v]);
      }
  }
  __syncthreads();

  // ---- 3/4. two MLP layers: H = tanh(H @ W^T + b) ----
#pragma unroll 1
  for (int layer = 0; layer < 2; ++layer) {
    const unsigned* Wp = layer ? W2p : W1p;
    const float* bias  = layer ? b2 : b1;
    const float bn0 = bias[nj0 * 16 + cn];
    const float bn1 = bias[(nj0 + 1) * 16 + cn];
#pragma unroll
    for (int t = 0; t < 8; ++t) acc[t] = vzero8();
#pragma unroll
    for (int kk = 0; kk < 8; ++kk) {
      Frag16 fb0, fb1;
      {
        const unsigned* p0 = Wp + (size_t)(((((kk << 4) + nj0) << 5) + lane)) * 8;
        const unsigned* p1 = Wp + (size_t)(((((kk << 4) + nj0 + 1) << 5) + lane)) * 8;
        fb0.q[0] = ((const uint4*)p0)[0]; fb0.q[1] = ((const uint4*)p0)[1];
        fb1.q[0] = ((const uint4*)p1)[0]; fb1.q[1] = ((const uint4*)p1)[1];
      }
      const int kb = (kk << 5) + ((lane & 16) ? 16 : 0);   // K base for this lane
#pragma unroll
      for (int mi = 0; mi < 4; ++mi) {
        const int row = mi * 16 + cn;
        Frag16 fa;
        fa.q[0] = *(const uint4*)&s_hrow[row * 128 + (kb >> 1)];
        fa.q[1] = *(const uint4*)&s_hrow[row * 128 + (kb >> 1) + 4];
        acc[mi * 2 + 0] = __builtin_amdgcn_wmma_f32_16x16x32_bf16(
            false, fa.v, false, fb0.v, (short)0, acc[mi * 2 + 0], false, false);
        acc[mi * 2 + 1] = __builtin_amdgcn_wmma_f32_16x16x32_bf16(
            false, fa.v, false, fb1.v, (short)0, acc[mi * 2 + 1], false, false);
      }
    }
    __syncthreads();   // all reads of s_hrow done before in-place overwrite

    if (layer == 0) {
      unsigned short* hb = (unsigned short*)s_hrow;
#pragma unroll
      for (int mi = 0; mi < 4; ++mi)
#pragma unroll
        for (int nj = 0; nj < 2; ++nj) {
          const int n  = (nj0 + nj) * 16 + cn;
          const int rb = mi * 16 + rsel;
          const float bn = nj ? bn1 : bn0;
          v8f a = acc[mi * 2 + nj];
#pragma unroll
          for (int v = 0; v < 8; ++v) hb[(rb + v) * DIM + n] = f2bf_tr(fast_tanh(a[v] + bn));
        }
      __syncthreads();
    } else {
      // final layer: tanh, then deterministic per-column partial sums.
      // After in-register reduction over mi and the 8 accumulator rows,
      // each column has exactly two lane contributions (lane l and l+16).
      float cs0 = 0.0f, cs1 = 0.0f;
#pragma unroll
      for (int mi = 0; mi < 4; ++mi) {
        v8f a0 = acc[mi * 2 + 0], a1 = acc[mi * 2 + 1];
#pragma unroll
        for (int v = 0; v < 8; ++v) {
          cs0 += fast_tanh(a0[v] + bn0);
          cs1 += fast_tanh(a1[v] + bn1);
        }
      }
      float* cpart = (float*)s_bpan;               // 2 x 256 floats (B-panel is dead)
      const int part = (lane >> 4);                // 0: rows 0..7 block, 1: rows 8..15 block
      cpart[part * DIM + nj0 * 16 + cn]       = cs0;
      cpart[part * DIM + (nj0 + 1) * 16 + cn] = cs1;
      __syncthreads();
      partial[(size_t)blk * DIM + tid] = cpart[tid] + cpart[DIM + tid];
    }
  }
}

// ---------------------------------------------------------------------------
// Finalize: pool over windows, /S, then 256 -> 2 head.  32 blocks x 256 thr.
// ---------------------------------------------------------------------------
__global__ __launch_bounds__(256) void finalize_kernel(
    const float* __restrict__ partial, const float* __restrict__ Wc,
    const float* __restrict__ bc, float* __restrict__ out) {
  __shared__ float red[256];
  const int b = blockIdx.x, d = threadIdx.x;
  float s = 0.0f;
#pragma unroll 4
  for (int w = 0; w < NWIN; ++w) s += partial[(size_t)((b * NWIN + w)) * DIM + d];
  const float pooled = s * (1.0f / (float)S_LEN);
#pragma unroll 1
  for (int c = 0; c < 2; ++c) {
    red[d] = pooled * Wc[c * DIM + d];
    __syncthreads();
    for (int off = 128; off > 0; off >>= 1) {
      if (d < off) red[d] += red[d + off];
      __syncthreads();
    }
    if (d == 0) out[b * 2 + c] = red[0] + bc[c];
    __syncthreads();
  }
}

// ---------------------------------------------------------------------------
extern "C" void kernel_launch(void* const* d_in, const int* in_sizes, int n_in,
                              void* d_out, int out_size, void* d_ws, size_t ws_size,
                              hipStream_t stream) {
  (void)in_sizes; (void)n_in; (void)out_size; (void)ws_size;
  const int*   x   = (const int*)d_in[0];
  const float* emb = (const float*)d_in[1];
  const float* W1  = (const float*)d_in[2];
  const float* b1  = (const float*)d_in[3];
  const float* W2  = (const float*)d_in[4];
  const float* b2  = (const float*)d_in[5];
  const float* Wc  = (const float*)d_in[6];
  const float* bc  = (const float*)d_in[7];

  unsigned* ws  = (unsigned*)d_ws;
  unsigned* Ap  = ws + A_OFF;
  unsigned* W1p = ws + W1_OFF;
  unsigned* W2p = ws + W2_OFF;
  float* partial = (float*)(ws + P_OFF);

  prep_a_kernel<<<NWIN, 256, 0, stream>>>(Ap);
  prep_w_kernel<<<128, 32, 0, stream>>>(W1, W1p);
  prep_w_kernel<<<128, 32, 0, stream>>>(W2, W2p);
  fused_attn_mlp<<<NBATCH * NWIN, 256, 0, stream>>>(x, emb, b1, b2, Ap, W1p, W2p, partial);
  finalize_kernel<<<NBATCH, 256, 0, stream>>>(partial, Wc, bc, (float*)d_out);
}
